// CustomBiLSTMModel_7490422964449
// MI455X (gfx1250) — matrix operand order repaired and checked by reading, not compiled, not run on hardware
//
#include <hip/hip_runtime.h>
#include <hip/hip_bf16.h>

typedef __attribute__((ext_vector_type(16))) __bf16 bf16x16;
typedef __attribute__((ext_vector_type(8)))  float  floatx8;
typedef __attribute__((ext_vector_type(4)))  int    v4i_;

#define BATCH 16384
#define DIN   512
#define HID   1024
#define OUTD  256
#define WROW  (DIN + HID)   // 1536: row stride of gate weight matrices
#define HBUF_COLS (2 * HID) // 2048
#define LDSPAD 8            // bf16 row padding to skew LDS banks

union FragU { bf16x16 v; uint4 q[2]; };

// Branch-free activations on hardware transcendentals (v_exp_f32 / v_rcp_f32).
#define LOG2E 1.44269504088896340736f
__device__ __forceinline__ float fast_sigmoid(float v) {
    float t = __builtin_amdgcn_exp2f(v * -LOG2E);
    return __builtin_amdgcn_rcpf(1.0f + t);
}
__device__ __forceinline__ float fast_tanh(float v) {
    float t = __builtin_amdgcn_exp2f(v * (2.0f * LOG2E));
    return 1.0f - 2.0f * __builtin_amdgcn_rcpf(1.0f + t);
}

// CDNA5 async global->LDS copy path (ASYNCcnt-tracked), sync fallback.
#if defined(__has_builtin)
#  if __has_builtin(__builtin_amdgcn_global_load_async_to_lds_b128) && \
      __has_builtin(__builtin_amdgcn_s_wait_asynccnt)
#    define HAVE_ASYNC_LDS 1
#  endif
#endif

#ifdef HAVE_ASYNC_LDS
typedef __attribute__((address_space(1))) v4i_* gptr_v4i;
typedef __attribute__((address_space(3))) v4i_* lptr_v4i;
#endif

__device__ __forceinline__ void copy16_g2l(const __bf16* src, __bf16* dstLds) {
#ifdef HAVE_ASYNC_LDS
    __builtin_amdgcn_global_load_async_to_lds_b128(
        (gptr_v4i)(const void*)src, (lptr_v4i)(void*)dstLds, 0, 0);
#else
    *(uint4*)dstLds = *(const uint4*)src;
#endif
}
__device__ __forceinline__ void async_wait_all() {
#ifdef HAVE_ASYNC_LDS
    __builtin_amdgcn_s_wait_asynccnt(0);
#endif
}

// ---------------------------------------------------------------------------
// Pre-pack: f32 [rows x src_stride] -> contiguous bf16 [rows x cols].
// One thread per 8 elements.
// ---------------------------------------------------------------------------
__global__ __launch_bounds__(256)
void cvt_f32_bf16_kernel(const float* __restrict__ src, __bf16* __restrict__ dst,
                         int rows, int src_stride, int cols) {
    const int idx = blockIdx.x * 256 + threadIdx.x;
    const int perRow = cols >> 3;
    const int row = idx / perRow;
    const int col = (idx - row * perRow) << 3;
    if (row >= rows) return;
    const float* s = src + (size_t)row * src_stride + col;
    float tmp[8];
    *(float4*)&tmp[0] = ((const float4*)s)[0];
    *(float4*)&tmp[4] = ((const float4*)s)[1];
    __bf16 o[8];
    #pragma unroll
    for (int i = 0; i < 8; ++i) o[i] = (__bf16)tmp[i];
    *(uint4*)(dst + (size_t)row * cols + col) = *(const uint4*)o;
}

// ---------------------------------------------------------------------------
// Kernel 1: fused i/o/g gate GEMMs + LSTM pointwise math for one direction.
// BM=128 x BN=64, BK=32. Operands pre-packed bf16; all staging is async
// global->LDS with asynccnt double buffering. 12 v_wmma per K-step.
// wpack = 3 gate matrices [3][HID][DIN] contiguous for this direction.
// ---------------------------------------------------------------------------
__global__ __launch_bounds__(256)
void bilstm_gates_kernel(const __bf16* __restrict__ xb,
                         const __bf16* __restrict__ wpack,
                         const float* __restrict__ bi,
                         const float* __restrict__ bo,
                         const float* __restrict__ bg,
                         __bf16* __restrict__ hbuf, int colOff) {
    __shared__ __align__(16) __bf16 As[2][128][32 + LDSPAD];
    __shared__ __align__(16) __bf16 Bs[2][3][64][32 + LDSPAD];

    const int t    = threadIdx.x;
    const int lane = t & 31;
    const int wv   = t >> 5;
    const int wm   = wv >> 2;          // 0..1
    const int wn   = wv & 3;           // 0..3
    const int lm   = lane & 15;
    const int lh   = lane >> 4;

    const int m0 = blockIdx.x * 128;
    const int n0 = blockIdx.y * 64;

    floatx8 acc[4][3] = {};            // [m-subtile][gate]

    auto stage = [&](int pb, int kk) {
        // A: 128x32 bf16 = 512 16B chunks, 2 per thread
        #pragma unroll
        for (int c = 0; c < 2; ++c) {
            const int chunk = c * 256 + t;
            const int row   = chunk >> 2;
            const int col   = (chunk & 3) << 3;
            copy16_g2l(xb + (size_t)(m0 + row) * DIN + kk + col,
                       &As[pb][row][col]);
        }
        // B: 3 x (64x32) bf16, 1 chunk per thread per gate
        const int row = t >> 2;
        const int col = (t & 3) << 3;
        #pragma unroll
        for (int g = 0; g < 3; ++g) {
            copy16_g2l(wpack + ((size_t)g * HID + n0 + row) * DIN + kk + col,
                       &Bs[pb][g][row][col]);
        }
    };

    stage(0, 0);                               // prologue: tile 0 in flight
    const int NIT = DIN / 32;                  // 16
    for (int it = 0; it < NIT; ++it) {
        const int pb = it & 1;
        async_wait_all();                      // tile `it` landed in LDS
        __syncthreads();
        if (it + 1 < NIT) stage(pb ^ 1, (it + 1) * 32);   // prefetch next

        FragU bf[3];
        #pragma unroll
        for (int g = 0; g < 3; ++g) {
            const uint4* p = (const uint4*)&Bs[pb][g][wn * 16 + lm][lh * 8];
            bf[g].q[0] = p[0];
            bf[g].q[1] = p[2];
        }
        #pragma unroll
        for (int ms = 0; ms < 4; ++ms) {
            const int arow = wm * 64 + ms * 16 + lm;
            FragU af;
            af.q[0] = *(const uint4*)&As[pb][arow][lh * 8];
            af.q[1] = *(const uint4*)&As[pb][arow][16 + lh * 8];
            #pragma unroll
            for (int g = 0; g < 3; ++g) {
                acc[ms][g] = __builtin_amdgcn_wmma_f32_16x16x32_bf16(
                    false, af.v, false, bf[g].v, (short)0, acc[ms][g], false, false);
            }
        }
        __syncthreads();                       // done reading buffer pb
    }

    // ---- epilogue: bias + activations + bf16 store of h ----
    const int ncol = n0 + wn * 16 + lm;
    const float biasI = bi[ncol];
    const float biasO = bo[ncol];
    const float biasG = bg[ncol];

    #pragma unroll
    for (int ms = 0; ms < 4; ++ms) {
        #pragma unroll
        for (int r = 0; r < 8; ++r) {
            float iv = fast_sigmoid(acc[ms][0][r] + biasI);
            float ov = fast_sigmoid(acc[ms][1][r] + biasO);
            float gv = fast_tanh(acc[ms][2][r] + biasG);
            float cv = iv * gv;               // f*c_prev == 0: f-gate dead
            float hv = ov * fast_tanh(cv);
            const int grow = m0 + wm * 64 + ms * 16 + r + lh * 8;  // C layout
            hbuf[(size_t)grow * HBUF_COLS + colOff + ncol] = (__bf16)hv;
        }
    }
}

// ---------------------------------------------------------------------------
// Kernel 2: out[B,256] = h[B,2048](bf16) @ Wd.T(bf16) + bd  (f32 out)
// BM=128 x BN=64, K = 2048 step 32, async double-buffered staging.
// ---------------------------------------------------------------------------
__global__ __launch_bounds__(256)
void bilstm_dense_kernel(const __bf16* __restrict__ hbuf,
                         const __bf16* __restrict__ wdb,
                         const float* __restrict__ bd,
                         float* __restrict__ out) {
    __shared__ __align__(16) __bf16 As[2][128][32 + LDSPAD];
    __shared__ __align__(16) __bf16 Bs[2][64][32 + LDSPAD];

    const int t    = threadIdx.x;
    const int lane = t & 31;
    const int wv   = t >> 5;
    const int wm   = wv >> 2;
    const int wn   = wv & 3;
    const int lm   = lane & 15;
    const int lh   = lane >> 4;

    const int m0 = blockIdx.x * 128;
    const int n0 = blockIdx.y * 64;

    floatx8 acc[4] = {};

    auto stage = [&](int pb, int kk) {
        #pragma unroll
        for (int c = 0; c < 2; ++c) {
            const int chunk = c * 256 + t;
            const int row   = chunk >> 2;
            const int col   = (chunk & 3) << 3;
            copy16_g2l(hbuf + (size_t)(m0 + row) * HBUF_COLS + kk + col,
                       &As[pb][row][col]);
        }
        const int row = t >> 2;
        const int col = (t & 3) << 3;
        copy16_g2l(wdb + (size_t)(n0 + row) * HBUF_COLS + kk + col,
                   &Bs[pb][row][col]);
    };

    stage(0, 0);
    const int NIT = HBUF_COLS / 32;            // 64
    for (int it = 0; it < NIT; ++it) {
        const int pb = it & 1;
        async_wait_all();
        __syncthreads();
        if (it + 1 < NIT) stage(pb ^ 1, (it + 1) * 32);

        FragU bf;
        {
            const uint4* p = (const uint4*)&Bs[pb][wn * 16 + lm][lh * 8];
            bf.q[0] = p[0];
            bf.q[1] = p[2];
        }
        #pragma unroll
        for (int ms = 0; ms < 4; ++ms) {
            const int arow = wm * 64 + ms * 16 + lm;
            FragU af;
            af.q[0] = *(const uint4*)&As[pb][arow][lh * 8];
            af.q[1] = *(const uint4*)&As[pb][arow][16 + lh * 8];
            acc[ms] = __builtin_amdgcn_wmma_f32_16x16x32_bf16(
                false, af.v, false, bf.v, (short)0, acc[ms], false, false);
        }
        __syncthreads();
    }

    const int ncol = n0 + wn * 16 + lm;
    const float bv = bd[ncol];
    #pragma unroll
    for (int ms = 0; ms < 4; ++ms) {
        #pragma unroll
        for (int r = 0; r < 8; ++r) {
            const int grow = m0 + wm * 64 + ms * 16 + r + lh * 8;
            out[(size_t)grow * OUTD + ncol] = acc[ms][r] + bv;
        }
    }
}

// ---------------------------------------------------------------------------
// Launch. Inputs: x, (Wi_f,bi_f),(Wf_f,bf_f),(Wo_f,bo_f),(Wg_f,bg_f),
//                 (Wi_b,bi_b),(Wf_b,bf_b),(Wo_b,bo_b),(Wg_b,bg_b), Wd, bd
// f-gate weights are dead (c_prev == 0). Workspace layout (bf16):
//   hbuf [16384,2048] 64MB | xb [16384,512] 16MB | wb [6,1024,512] 6MB
//   | wdb [256,2048] 1MB   -> ~87 MB total
// ---------------------------------------------------------------------------
extern "C" void kernel_launch(void* const* d_in, const int* in_sizes, int n_in,
                              void* d_out, int out_size, void* d_ws, size_t ws_size,
                              hipStream_t stream) {
    const float* x    = (const float*)d_in[0];
    const float* Wg6[6] = {
        (const float*)d_in[1],  // Wi_f
        (const float*)d_in[5],  // Wo_f
        (const float*)d_in[7],  // Wg_f
        (const float*)d_in[9],  // Wi_b
        (const float*)d_in[13], // Wo_b
        (const float*)d_in[15], // Wg_b
    };
    const float* bi_f = (const float*)d_in[2];
    const float* bo_f = (const float*)d_in[6];
    const float* bg_f = (const float*)d_in[8];
    const float* bi_b = (const float*)d_in[10];
    const float* bo_b = (const float*)d_in[14];
    const float* bg_b = (const float*)d_in[16];
    const float* Wd   = (const float*)d_in[17];
    const float* bd   = (const float*)d_in[18];
    float* out = (float*)d_out;

    char* ws = (char*)d_ws;
    __bf16* hbuf = (__bf16*)ws;
    __bf16* xb   = (__bf16*)(ws + (size_t)BATCH * HBUF_COLS * sizeof(__bf16));
    __bf16* wb   = xb + (size_t)BATCH * DIN;
    __bf16* wdb  = wb + (size_t)6 * HID * DIN;

    dim3 blk(256, 1, 1);

    // ---- pre-pack to bf16 ----
    cvt_f32_bf16_kernel<<<(BATCH * DIN / 8) / 256, blk, 0, stream>>>(
        x, xb, BATCH, DIN, DIN);
    for (int g = 0; g < 6; ++g) {
        cvt_f32_bf16_kernel<<<(HID * DIN / 8) / 256, blk, 0, stream>>>(
            Wg6[g], wb + (size_t)g * HID * DIN, HID, WROW, DIN);
    }
    cvt_f32_bf16_kernel<<<(OUTD * HBUF_COLS / 8) / 256, blk, 0, stream>>>(
        Wd, wdb, OUTD, HBUF_COLS, HBUF_COLS);

    // ---- GEMMs ----
    dim3 grdGates(BATCH / 128, HID / 64, 1);   // 128 x 16
    dim3 grdDense(BATCH / 128, OUTD / 64, 1);  // 128 x 4

    bilstm_gates_kernel<<<grdGates, blk, 0, stream>>>(
        xb, wb, bi_f, bo_f, bg_f, hbuf, 0);
    bilstm_gates_kernel<<<grdGates, blk, 0, stream>>>(
        xb, wb + (size_t)3 * HID * DIN, bi_b, bo_b, bg_b, hbuf, HID);
    bilstm_dense_kernel<<<grdDense, blk, 0, stream>>>(hbuf, wdb, bd, out);
}
